// MappingLayer_75685913690508
// MI455X (gfx1250) — matrix operand order repaired and checked by reading, not compiled, not run on hardware
//
#include <hip/hip_runtime.h>
#include <hip/hip_bf16.h>
#include <math.h>

// Problem constants (from reference): B=64, A=128, IN_C=128, OUT_C=64
// N = B*A = 8192 tokens, d = 64.
#define NTOK   8192
#define DHEAD  64
#define INC    128
#define NB     64
#define AA     128

typedef _Float16 v16h __attribute__((ext_vector_type(16)));
typedef _Float16 v8h  __attribute__((ext_vector_type(8)));
typedef float    v8f  __attribute__((ext_vector_type(8)));

union H16 {
    v16h v;
    v8h  h8[2];
    unsigned int w[8];
    _Float16 h[16];
};
union PK2 {
    unsigned int w;
    _Float16 h[2];
};

// exp2 that lowers to a single v_exp_f32 (scale folded by caller).
#if __has_builtin(__builtin_amdgcn_exp2f)
#define FAST_EXP2(x) __builtin_amdgcn_exp2f(x)
#else
#define FAST_EXP2(x) exp2f(x)
#endif

// 1/sqrt(d) * log2(e): single pre-scale for exp2-based softmax.
#define SCALE_LOG2E 0.1803368801111244f

static __device__ __forceinline__ v8f wmma_f16(v16h a, v16h b, v8f c) {
    // D = A(16x32 f16) x B(32x16 f16) + C(16x16 f32)
    return __builtin_amdgcn_wmma_f32_16x16x32_f16(
        /*neg_a=*/false, a, /*neg_b=*/false, b,
        /*c_mod=*/(short)0, c, /*reuse_a=*/false, /*reuse_b=*/false);
}

// ---------------------------------------------------------------------------
// Kernel 1: QKV projection.
//   f[n, ic] = features[b, ic, a], n = b*128 + a
//   q = f Wq^T + bq  (etc.)  Emits: qf16 (N x 64), kf16 (N x 64),
//   vT16 (64 x N, transposed for WMMA A-fragments), v_n f32 (N x 64, L2-normed).
// Block = 256 threads = 4 rows x 64 output channels.
// ---------------------------------------------------------------------------
__global__ __launch_bounds__(256) void qkv_proj(
    const float* __restrict__ feat,
    const float* __restrict__ Wq, const float* __restrict__ bq,
    const float* __restrict__ Wk, const float* __restrict__ bk,
    const float* __restrict__ Wv, const float* __restrict__ bv,
    _Float16* __restrict__ qf, _Float16* __restrict__ kf,
    _Float16* __restrict__ vT, float* __restrict__ vn)
{
    __shared__ float fs[4][INC];
    __shared__ float red[256];

    const int n0 = blockIdx.x * 4;          // 4 rows per block, all in same batch b
    const int b  = n0 >> 7;
    const int a0 = n0 & 127;
    const int t  = threadIdx.x;

    // Stage 4 f-rows (gather the feature transpose) into LDS.
    for (int i = t; i < 4 * INC; i += 256) {
        const int rr = i & 3, ic = i >> 2;
        fs[rr][ic] = feat[b * (INC * AA) + ic * AA + a0 + rr];
    }
    __syncthreads();

    const int rr = t >> 6;                  // row within block (0..3)
    const int j  = t & 63;                  // output channel (0..63)

    const float* __restrict__ wq = Wq + j * INC;
    const float* __restrict__ wk = Wk + j * INC;
    const float* __restrict__ wv = Wv + j * INC;

    float aq = bq[j], ak = bk[j], av = bv[j];
#pragma unroll 8
    for (int ic = 0; ic < INC; ++ic) {
        const float x = fs[rr][ic];
        aq = fmaf(x, wq[ic], aq);
        ak = fmaf(x, wk[ic], ak);
        av = fmaf(x, wv[ic], av);
    }

    // L2 norm of the v row (64 channels spread across threads with same rr).
    red[t] = av * av;
    __syncthreads();
    float n2 = 0.f;
#pragma unroll 8
    for (int jj = 0; jj < 64; ++jj) n2 += red[rr * 64 + jj];
    const float scale = 1.0f / fmaxf(sqrtf(n2), 1e-12f);

    const int n = n0 + rr;
    qf[n * DHEAD + j] = (_Float16)aq;
    kf[n * DHEAD + j] = (_Float16)ak;
    vT[j * NTOK + n]  = (_Float16)av;       // transposed for the PV A-fragments
    vn[n * DHEAD + j] = av * scale;
}

// ---------------------------------------------------------------------------
// Kernel 2: streaming (flash) attention + fused L2 normalize.
// Block = 4 waves; all 4 waves share the SAME 16 queries, each handling a
// 2048-key quarter of the key range (sum-only softmax is associative, so the
// split is exact and deterministic). Per 32-key chunk per wave:
//   S^T (32 keys x 16 q) via 4 WMMA:  A = K-tile rows, B = Q^T.
//   P^T -> B-fragment with 8 shfl_xor(16).
//   O^T += V^T x P^T via 4 WMMA (A = V^T dim-rows, contiguous loads).
// Partial (O^T, sum_exp) combined through padded LDS in fixed order; wave 0
// divides by the row sum, L2-normalizes, and writes qkv_n (N x 64 f32).
// ---------------------------------------------------------------------------
__global__ __launch_bounds__(128) void flash_attn(
    const _Float16* __restrict__ qf,
    const _Float16* __restrict__ kf,
    const _Float16* __restrict__ vT,
    float* __restrict__ qkvn)
{
    const int lane = threadIdx.x & 31;
    const int wid  = threadIdx.x >> 5;      // 0..3: key-range quarter
    const int q0   = blockIdx.x * 16;       // 16 queries per block
    const bool hi  = lane >= 16;
    const int col  = lane & 15;             // query column of this lane
    const int klo  = hi ? 8 : 0;            // A-fragment K-offset for this half-wave

    // Q^T B-fragments (loaded once): lane holds halfs klo..klo+7 and klo+16..klo+23 of each 32-dim slice of its query row.
    H16 bq0, bq1;
    {
        const _Float16* qrow = qf + (q0 + col) * DHEAD;
        bq0.h8[0] = *(const v8h*)(qrow + klo);        // dims 0..31
        bq0.h8[1] = *(const v8h*)(qrow + 16 + klo);
        bq1.h8[0] = *(const v8h*)(qrow + 32 + klo);   // dims 32..63
        bq1.h8[1] = *(const v8h*)(qrow + 48 + klo);
    }

    v8f o[4] = {v8f{}, v8f{}, v8f{}, v8f{}};   // O^T accumulators (64 dims x 16 q)
    float lsum = 0.f;

    const int k_begin = wid * (NTOK / 4);
    const int k_end   = k_begin + (NTOK / 4);

    for (int kc = k_begin; kc < k_end; kc += 32) {
        // Prefetch next chunk (K rows are one 128B line each; V^T spans).
        __builtin_prefetch(kf + (kc + 32 + col) * DHEAD, 0, 3);
        __builtin_prefetch(kf + (kc + 48 + col) * DHEAD, 0, 3);
#pragma unroll
        for (int jj = 0; jj < 4; ++jj)
            __builtin_prefetch(vT + (16 * jj + col) * NTOK + kc + 32, 0, 3);

        // --- S^T = K * Q^T -----------------------------------------------
        H16 ak00, ak01, ak10, ak11;
        {
            const _Float16* kr0 = kf + (kc + col) * DHEAD;        // keys kc..kc+15
            const _Float16* kr1 = kf + (kc + 16 + col) * DHEAD;   // keys kc+16..kc+31
            ak00.h8[0] = *(const v8h*)(kr0 + klo);
            ak00.h8[1] = *(const v8h*)(kr0 + 16 + klo);
            ak01.h8[0] = *(const v8h*)(kr0 + 32 + klo);
            ak01.h8[1] = *(const v8h*)(kr0 + 48 + klo);
            ak10.h8[0] = *(const v8h*)(kr1 + klo);
            ak10.h8[1] = *(const v8h*)(kr1 + 16 + klo);
            ak11.h8[0] = *(const v8h*)(kr1 + 32 + klo);
            ak11.h8[1] = *(const v8h*)(kr1 + 48 + klo);
        }
        v8f c0 = {}, c1 = {};
        c0 = wmma_f16(ak00.v, bq0.v, c0);
        c0 = wmma_f16(ak01.v, bq1.v, c0);
        c1 = wmma_f16(ak10.v, bq0.v, c1);
        c1 = wmma_f16(ak11.v, bq1.v, c1);

        // --- exp2(s * (log2e/sqrt(d))): one mul + one v_exp per element ---
        unsigned int w0[4], w1[4];
#pragma unroll
        for (int k = 0; k < 4; ++k) {
            PK2 p;
            float e0 = FAST_EXP2(c0[2 * k] * SCALE_LOG2E);
            float e1 = FAST_EXP2(c0[2 * k + 1] * SCALE_LOG2E);
            lsum += e0 + e1;
            p.h[0] = (_Float16)e0; p.h[1] = (_Float16)e1;
            w0[k] = p.w;
            e0 = FAST_EXP2(c1[2 * k] * SCALE_LOG2E);
            e1 = FAST_EXP2(c1[2 * k + 1] * SCALE_LOG2E);
            lsum += e0 + e1;
            p.h[0] = (_Float16)e0; p.h[1] = (_Float16)e1;
            w1[k] = p.w;
        }
        // C-layout -> B-layout for P^T: exchange halves between lane pairs.
        unsigned int ow0[4], ow1[4];
#pragma unroll
        for (int k = 0; k < 4; ++k) {
            ow0[k] = (unsigned int)__shfl_xor((int)w0[k], 16, 32);
            ow1[k] = (unsigned int)__shfl_xor((int)w1[k], 16, 32);
        }
        H16 bp;
#pragma unroll
        for (int k = 0; k < 4; ++k) {
            bp.w[k]     = hi ? ow1[k] : w0[k];  // K = base+0..7
            bp.w[4 + k] = hi ? w1[k]  : ow0[k]; // K = base+8..15
        }

        // --- O^T += V^T * P^T --------------------------------------------
#pragma unroll
        for (int jj = 0; jj < 4; ++jj) {
            H16 av;
            const _Float16* vr = vT + (16 * jj + col) * NTOK + kc;
            av.v = *(const v16h*)(vr + (hi ? 16 : 0));
            o[jj] = wmma_f16(av.v, bp.v, o[jj]);
        }
    }

    // --- Cross-wave combine (deterministic fixed-order sums) --------------
    __shared__ float po[4][32][33];   // padded: stride 33 -> conflict-free
    __shared__ float pl[4][32];

#pragma unroll
    for (int jj = 0; jj < 4; ++jj)
#pragma unroll
        for (int r = 0; r < 8; ++r)
            po[wid][lane][jj * 8 + r] = o[jj][r];
    pl[wid][lane] = lsum;
    __syncthreads();

    if (wid == 0) {
        float ls = 0.f;
#pragma unroll
        for (int w = 0; w < 4; ++w) ls += pl[w][lane];

        float ot[4][8];
#pragma unroll
        for (int jj = 0; jj < 4; ++jj)
#pragma unroll
            for (int r = 0; r < 8; ++r) {
                const int i = jj * 8 + r;
                ot[jj][r] = ((po[0][lane][i] + po[1][lane][i]) +
                             (po[2][lane][i] + po[3][lane][i]));
            }

        // Softmax denominator, then fused L2 normalize of the O row.
        const float l    = ls + __shfl_xor(ls, 16, 32);
        const float invl = 1.0f / l;

        float vals[4][8];
        float ss = 0.f;
#pragma unroll
        for (int jj = 0; jj < 4; ++jj)
#pragma unroll
            for (int r = 0; r < 8; ++r) {
                const float v = ot[jj][r] * invl;
                vals[jj][r] = v;
                ss += v * v;
            }
        ss += __shfl_xor(ss, 16, 32);             // full 64-dim sum for query `col`
        const float nscale = 1.0f / fmaxf(sqrtf(ss), 1e-12f);

        const int qrow = q0 + col;
#pragma unroll
        for (int jj = 0; jj < 4; ++jj)
#pragma unroll
            for (int r = 0; r < 8; ++r) {
                const int dim = 16 * jj + r + klo;
                qkvn[qrow * DHEAD + dim] = vals[jj][r] * nscale;
            }
    }
}

// ---------------------------------------------------------------------------
// Kernel 3: sim_s[b,c] = (1/A) * <v_n batch-row b, qkv_n batch-row c>
// Batch-rows are contiguous 8192-float spans. One block per b; v cached in
// registers; loop over the 64 c rows with a block reduction.
// ---------------------------------------------------------------------------
__global__ __launch_bounds__(256) void sim64(
    const float* __restrict__ vn, const float* __restrict__ qn,
    float* __restrict__ out)
{
    const int b = blockIdx.x;
    const int t = threadIdx.x;
    const int lane = t & 31, wid = t >> 5;

    float vreg[32];
#pragma unroll
    for (int k = 0; k < 32; ++k)
        vreg[k] = vn[b * (AA * DHEAD) + t + 256 * k];

    __shared__ float red[8];

    for (int c = 0; c < NB; ++c) {
        const float* qr = qn + c * (AA * DHEAD);
        float acc = 0.f;
#pragma unroll
        for (int k = 0; k < 32; ++k)
            acc = fmaf(vreg[k], qr[t + 256 * k], acc);
#pragma unroll
        for (int off = 16; off >= 1; off >>= 1)
            acc += __shfl_xor(acc, off, 32);
        if (lane == 0) red[wid] = acc;
        __syncthreads();
        if (t == 0) {
            float s = 0.f;
#pragma unroll
            for (int w = 0; w < 8; ++w) s += red[w];
            out[b * NB + c] = s * (1.0f / (float)AA);
        }
        __syncthreads();
    }
}

// ---------------------------------------------------------------------------
extern "C" void kernel_launch(void* const* d_in, const int* in_sizes, int n_in,
                              void* d_out, int out_size, void* d_ws, size_t ws_size,
                              hipStream_t stream) {
    (void)in_sizes; (void)n_in; (void)out_size; (void)ws_size;

    const float* feat = (const float*)d_in[0];
    const float* Wq   = (const float*)d_in[1];
    const float* bq   = (const float*)d_in[2];
    const float* Wk   = (const float*)d_in[3];
    const float* bk   = (const float*)d_in[4];
    const float* Wv   = (const float*)d_in[5];
    const float* bv   = (const float*)d_in[6];
    float* out        = (float*)d_out;

    // Workspace layout (~7.3 MB total)
    char* ws = (char*)d_ws;
    _Float16* qf16 = (_Float16*)ws;                  // 8192*64*2 = 1 MB
    _Float16* kf16 = qf16 + NTOK * DHEAD;            // 1 MB
    _Float16* vT16 = kf16 + NTOK * DHEAD;            // 1 MB (transposed V)
    float*    vn   = (float*)(vT16 + NTOK * DHEAD);  // 2 MB
    float*    qn   = vn + NTOK * DHEAD;              // 2 MB

    // 1) QKV projection: 8192 rows, 4 rows per 256-thread block.
    qkv_proj<<<NTOK / 4, 256, 0, stream>>>(feat, Wq, bq, Wk, bk, Wv, bv,
                                           qf16, kf16, vT16, vn);

    // 2) Flash attention + fused normalize: 512 blocks x 4 waves,
    //    each block owns 16 queries, each wave a quarter of the keys.
    flash_attn<<<NTOK / 16, 128, 0, stream>>>(qf16, kf16, vT16, qn);

    // 3) Final 64x64 similarity.
    sim64<<<NB, 256, 0, stream>>>(vn, qn, out);
}
